// MGMCGCN_69406671503625
// MI455X (gfx1250) — matrix operand order repaired and checked by reading, not compiled, not run on hardware
//
#include <hip/hip_runtime.h>

#define NN 4096
#define FD 256
#define HD 256
#define KK 3
#define TT 5

typedef __attribute__((ext_vector_type(16))) __bf16 v16bf;
typedef __attribute__((ext_vector_type(8)))  float  v8f;

union Frag { v16bf v; unsigned int u[8]; unsigned short s[16]; };

__device__ __forceinline__ unsigned short f2bf(float f) {
  unsigned int u = __float_as_uint(f);
  u += 0x7FFFu + ((u >> 16) & 1u);          // round-to-nearest-even
  return (unsigned short)(u >> 16);
}
__device__ __forceinline__ float sigm(float x) { return 1.0f / (1.0f + __expf(-x)); }

// ---------------------------------------------------------------------------
// Single dense pass over adj: one block per row i; thread t streams the
// contiguous 96-dword span covering j in [t*32, t*32+32) for ALL 3 k's.
// Emits: float copy to output, 0/1 bitmask per k, dinv per (k, i).
// Every byte of adj is read and written exactly once (int4/float4 streams).
// ---------------------------------------------------------------------------
__global__ __launch_bounds__(128)
void prep_adj(const int* __restrict__ adj, float* __restrict__ out_adj,
              unsigned int* __restrict__ abits, float* __restrict__ dinv) {
  const int i = blockIdx.x, t = threadIdx.x;
  __shared__ int cnt[KK];
  if (t < KK) cnt[t] = 0;
  __syncthreads();
  const int j0 = t * 32;                                   // 32 j's * 3 k's = 96 dwords
  const int4* src = (const int4*)(adj + ((size_t)i * NN + j0) * KK);
  float4*     dst = (float4*)(out_adj + ((size_t)i * NN + j0) * KK);
  unsigned int bits[KK] = {0u, 0u, 0u};
  #pragma unroll
  for (int c = 0; c < 24; ++c) {
    int4 v = src[c];
    float4 f;
    f.x = (float)v.x; f.y = (float)v.y; f.z = (float)v.z; f.w = (float)v.w;
    dst[c] = f;
    const int q = c * 4;                                   // dword q -> (j=q/3, k=q%3)
    bits[(q    ) % KK] |= (v.x != 0 ? 1u : 0u) << ((q    ) / KK);
    bits[(q + 1) % KK] |= (v.y != 0 ? 1u : 0u) << ((q + 1) / KK);
    bits[(q + 2) % KK] |= (v.z != 0 ? 1u : 0u) << ((q + 2) / KK);
    bits[(q + 3) % KK] |= (v.w != 0 ? 1u : 0u) << ((q + 3) / KK);
  }
  #pragma unroll
  for (int k = 0; k < KK; ++k) {
    abits[((size_t)k * NN + i) * (NN / 32) + t] = bits[k];
    atomicAdd(&cnt[k], __popc(bits[k]));
  }
  __syncthreads();
  if (t < KK) dinv[t * NN + i] = rsqrtf((float)(cnt[t] + 1));  // +1 self loop
}

// ---------------------------------------------------------------------------
// f32 -> bf16 convert (plain) and convert+transpose (dst[k][c][r]=src[k][r][c])
// ---------------------------------------------------------------------------
__global__ void cvt(const float* __restrict__ s, unsigned short* __restrict__ d, int n) {
  int i = blockIdx.x * blockDim.x + threadIdx.x;
  if (i < n) d[i] = f2bf(s[i]);
}
__global__ void cvtT(const float* __restrict__ s, unsigned short* __restrict__ d, int R, int C) {
  int i = blockIdx.x * blockDim.x + threadIdx.x;
  if (i >= R * C) return;
  int k = blockIdx.z;
  int r = i / C, c = i % C;
  d[(size_t)k * R * C + (size_t)c * R + r] = f2bf(s[(size_t)k * R * C + (size_t)r * C + c]);
}

// ---------------------------------------------------------------------------
// Generic bf16 GEMM: C[M,N] = A[M,Kd] @ B[Kd,N], all row-major, f32 out.
// 256 threads = 8 waves (4x2); BM=128 BN=64 BK=32; wave computes 32x32.
// Double-buffered LDS, one barrier per K-step; A tile staged via
// GLOBAL_LOAD_ASYNC_TO_LDS_B128 (ASYNCcnt), B transposed in registers.
// A frags read as 2x ds_load_b128 (80B row stride); B frags as paired b32.
// M % 128 == 0, N % 64 == 0, Kd % 32 == 0 (true for all uses here).
// ---------------------------------------------------------------------------
#define BM 128
#define BN 64
#define BK 32
#define APAD 8   // A row = 40 shorts = 80B: 16B-aligned chunks, 20-dword stride
#define BPAD 2   // B row = 34 shorts = 68B: 17-dword stride (conflict-free)

__global__ __launch_bounds__(256)
void gemm_bf16(const unsigned short* __restrict__ A, const unsigned short* __restrict__ B,
               float* __restrict__ C, int M, int N, int Kd) {
  __shared__ __align__(16) unsigned short As[2][BM][BK + APAD];
  __shared__ __align__(16) unsigned short Bs[2][BN][BK + BPAD];
  const int tid  = threadIdx.x;
  const int lane = tid & 31, wid = tid >> 5;
  const int wm = wid & 3, wn = wid >> 2;      // 4x2 wave grid
  const int bm = blockIdx.x * BM, bn = blockIdx.y * BN;
  const int m = lane & 15, h = lane >> 4;

  // A tile = 512 chunks of 16B; this thread owns chunks tid and tid+256.
  const int ar0 = tid >> 2, ac0 = (tid & 3) * 8;          // row, short-offset
  const int ar1 = (tid + 256) >> 2, ac1 = ac0;

  auto issueA = [&](int kk, int bufi) {
    const unsigned short* g0 = A + (size_t)(bm + ar0) * Kd + kk + ac0;
    const unsigned short* g1 = A + (size_t)(bm + ar1) * Kd + kk + ac1;
    unsigned l0 = (unsigned)(size_t)&As[bufi][ar0][ac0];
    unsigned l1 = (unsigned)(size_t)&As[bufi][ar1][ac1];
    asm volatile("global_load_async_to_lds_b128 %0, %1, off"
                 :: "v"(l0), "v"(g0) : "memory");
    asm volatile("global_load_async_to_lds_b128 %0, %1, off"
                 :: "v"(l1), "v"(g1) : "memory");
  };
  // B tile = 512 (row-pair, dword-col) units; pack two rows into dword stores.
  auto issueB = [&](int kk, int bufi) {
    #pragma unroll
    for (int it = 0; it < 2; ++it) {
      int d = tid + it * 256;
      int r = (d >> 5) * 2, c2 = d & 31;
      const unsigned short* gp = B + (size_t)(kk + r) * N + bn + c2 * 2;
      unsigned int v0 = *(const unsigned int*)gp;
      unsigned int v1 = *(const unsigned int*)(gp + N);
      unsigned int lo = (v0 & 0xffffu) | (v1 << 16);
      unsigned int hi = (v0 >> 16) | (v1 & 0xffff0000u);
      *(unsigned int*)&Bs[bufi][c2 * 2][r]     = lo;
      *(unsigned int*)&Bs[bufi][c2 * 2 + 1][r] = hi;
    }
  };

  v8f acc[2][2] = {};
  issueA(0, 0);
  issueB(0, 0);
  int buf = 0;
  for (int kk = 0; kk < Kd; kk += BK) {
    asm volatile("s_wait_asynccnt 0x0" ::: "memory");
    __syncthreads();
    if (kk + BK < Kd) { issueA(kk + BK, buf ^ 1); issueB(kk + BK, buf ^ 1); }

    Frag a[2], b[2];
    #pragma unroll
    for (int sm = 0; sm < 2; ++sm) {
      const unsigned int* r32 = (const unsigned int*)&As[buf][wm * 32 + sm * 16 + m][0];
      #pragma unroll
      for (int p = 0; p < 8; ++p) {
        int kd = (p < 4) ? (4 * h + p) : (4 * h + p + 4);  // ISA A-frag K interleave
        a[sm].u[p] = r32[kd];
      }
    }
    #pragma unroll
    for (int sn = 0; sn < 2; ++sn) {
      const unsigned int* c32 = (const unsigned int*)&Bs[buf][wn * 32 + sn * 16 + m][0];
      #pragma unroll
      for (int p = 0; p < 8; ++p) b[sn].u[p] = c32[(p < 4) ? (4 * h + p) : (4 * h + p + 4)];  // ISA K interleave (matches A)
    }
    #pragma unroll
    for (int sm = 0; sm < 2; ++sm)
      #pragma unroll
      for (int sn = 0; sn < 2; ++sn)
        acc[sm][sn] = __builtin_amdgcn_wmma_f32_16x16x32_bf16(
            false, a[sm].v, false, b[sn].v, (short)0, acc[sm][sn], false, false);
    buf ^= 1;
  }
  #pragma unroll
  for (int sm = 0; sm < 2; ++sm)
    #pragma unroll
    for (int sn = 0; sn < 2; ++sn) {
      int gm0 = bm + wm * 32 + sm * 16 + 8 * h;
      int gn  = bn + wn * 32 + sn * 16 + m;
      float* cp = C + (size_t)gm0 * N + gn;
      #pragma unroll
      for (int r = 0; r < 8; ++r) cp[(size_t)r * N] = acc[sm][sn][r];
    }
}

// ---------------------------------------------------------------------------
// Adjacency GEMM: C[NN,N] = Abin(bitmask) @ B[NN,N]; A frag built from bits
// with VALU selects (bf16 1.0/0.0), B double-buffered in LDS.
// ---------------------------------------------------------------------------
__global__ __launch_bounds__(256)
void gemm_adj(const unsigned int* __restrict__ Ab, const unsigned short* __restrict__ B,
              float* __restrict__ C, int N, int Kd) {
  __shared__ __align__(16) unsigned short Bs[2][BN][BK + BPAD];
  const int tid  = threadIdx.x;
  const int lane = tid & 31, wid = tid >> 5;
  const int wm = wid & 3, wn = wid >> 2;
  const int bm = blockIdx.x * BM, bn = blockIdx.y * BN;
  const int m = lane & 15, h = lane >> 4;
  const int kw = Kd >> 5;                     // dwords per bit-row

  auto issueB = [&](int kk, int bufi) {
    #pragma unroll
    for (int it = 0; it < 2; ++it) {
      int d = tid + it * 256;
      int r = (d >> 5) * 2, c2 = d & 31;
      const unsigned short* gp = B + (size_t)(kk + r) * N + bn + c2 * 2;
      unsigned int v0 = *(const unsigned int*)gp;
      unsigned int v1 = *(const unsigned int*)(gp + N);
      unsigned int lo = (v0 & 0xffffu) | (v1 << 16);
      unsigned int hi = (v0 >> 16) | (v1 & 0xffff0000u);
      *(unsigned int*)&Bs[bufi][c2 * 2][r]     = lo;
      *(unsigned int*)&Bs[bufi][c2 * 2 + 1][r] = hi;
    }
  };

  v8f acc[2][2] = {};
  issueB(0, 0);
  int buf = 0;
  for (int kk = 0; kk < Kd; kk += BK) {
    __syncthreads();
    if (kk + BK < Kd) issueB(kk + BK, buf ^ 1);

    Frag a[2], b[2];
    #pragma unroll
    for (int sm = 0; sm < 2; ++sm) {
      unsigned int w = Ab[(size_t)(bm + wm * 32 + sm * 16 + m) * kw + (kk >> 5)];
      #pragma unroll
      for (int e = 0; e < 16; ++e) {
        int bit = (e < 8) ? (8 * h + e) : (16 + 8 * h + (e - 8));
        a[sm].s[e] = ((w >> bit) & 1u) ? (unsigned short)0x3F80u : (unsigned short)0u;
      }
    }
    #pragma unroll
    for (int sn = 0; sn < 2; ++sn) {
      const unsigned int* c32 = (const unsigned int*)&Bs[buf][wn * 32 + sn * 16 + m][0];
      #pragma unroll
      for (int p = 0; p < 8; ++p) b[sn].u[p] = c32[(p < 4) ? (4 * h + p) : (4 * h + p + 4)];
    }
    #pragma unroll
    for (int sm = 0; sm < 2; ++sm)
      #pragma unroll
      for (int sn = 0; sn < 2; ++sn)
        acc[sm][sn] = __builtin_amdgcn_wmma_f32_16x16x32_bf16(
            false, a[sm].v, false, b[sn].v, (short)0, acc[sm][sn], false, false);
    buf ^= 1;
  }
  #pragma unroll
  for (int sm = 0; sm < 2; ++sm)
    #pragma unroll
    for (int sn = 0; sn < 2; ++sn) {
      int gm0 = bm + wm * 32 + sm * 16 + 8 * h;
      int gn  = bn + wn * 32 + sn * 16 + m;
      float* cp = C + (size_t)gm0 * N + gn;
      #pragma unroll
      for (int r = 0; r < 8; ++r) cp[(size_t)r * N] = acc[sm][sn][r];
    }
}

// ---------------------------------------------------------------------------
// Pointwise epilogues
// ---------------------------------------------------------------------------
__global__ void e1_scale(const float* __restrict__ xw, const float* __restrict__ dinv_k,
                         unsigned short* __restrict__ ybf) {
  int idx = blockIdx.x * blockDim.x + threadIdx.x;   // over NN*HD
  int n = idx >> 8;
  ybf[idx] = f2bf(dinv_k[n] * xw[idx]);
}

__global__ void e2_gcn(const float* __restrict__ z, const float* __restrict__ xw,
                       const float* __restrict__ dinv_k, const float* __restrict__ gnn_b_k,
                       unsigned short* __restrict__ gbf) {
  int idx = blockIdx.x * blockDim.x + threadIdx.x;
  int n = idx >> 8, hh = idx & 255;
  float di = dinv_k[n];
  float g = di * (z[idx] + di * xw[idx]) + gnn_b_k[hh];   // dinv*(A@Y + Y) + b
  gbf[idx] = f2bf(g);
}

__global__ void lstm_pw(const float* __restrict__ xg, const float* __restrict__ hg,
                        const float* __restrict__ b_ih_k, const float* __restrict__ b_hh_k,
                        float* __restrict__ c, unsigned short* __restrict__ hbf, int first) {
  int idx = blockIdx.x * blockDim.x + threadIdx.x;   // over NN*HD
  int n = idx >> 8, hh = idx & 255;
  size_t base = (size_t)n * (4 * HD);
  float gi = xg[base + hh]            + b_ih_k[hh]            + b_hh_k[hh];
  float gf = xg[base + HD + hh]       + b_ih_k[HD + hh]       + b_hh_k[HD + hh];
  float gg = xg[base + 2 * HD + hh]   + b_ih_k[2 * HD + hh]   + b_hh_k[2 * HD + hh];
  float go = xg[base + 3 * HD + hh]   + b_ih_k[3 * HD + hh]   + b_hh_k[3 * HD + hh];
  float cp = 0.0f;
  if (!first) {
    gi += hg[base + hh]; gf += hg[base + HD + hh];
    gg += hg[base + 2 * HD + hh]; go += hg[base + 3 * HD + hh];
    cp = c[idx];
  }
  float cn = sigm(gf) * cp + sigm(gi) * tanhf(gg);
  float hn = sigm(go) * tanhf(cn);
  c[idx] = cn;
  hbf[idx] = f2bf(hn);
}

__global__ void e3_fc(const float* __restrict__ dpre, const float* __restrict__ fc_b_k,
                      const float* __restrict__ x, float* __restrict__ dsum,
                      float* __restrict__ out_z, int k) {
  int idx = blockIdx.x * blockDim.x + threadIdx.x;   // over NN*FD
  int f = idx & 255;
  float delta = tanhf(dpre[idx] + fc_b_k[f]);
  float s = (k == 0) ? delta : (dsum[idx] + delta);
  dsum[idx] = s;
  float zb = x[idx] + delta;
  if (f >= 10 && f < FD - 3) zb = sigm(zb);
  out_z[(size_t)idx * KK + k] = zb;
}

__global__ void e4_pred(const float* __restrict__ x, const float* __restrict__ dsum,
                        float* __restrict__ out) {
  int idx = blockIdx.x * blockDim.x + threadIdx.x;   // over NN*FD
  int f = idx & 255;
  float v = x[idx] + dsum[idx] * (1.0f / 3.0f);
  if (f >= 10 && f < FD - 3) v = sigm(v);
  out[idx] = v;
}

// ---------------------------------------------------------------------------
extern "C" void kernel_launch(void* const* d_in, const int* in_sizes, int n_in,
                              void* d_out, int out_size, void* d_ws, size_t ws_size,
                              hipStream_t stream) {
  const float* x     = (const float*)d_in[0];
  const float* gnn_w = (const float*)d_in[1];
  const float* gnn_b = (const float*)d_in[2];
  const float* w_ih  = (const float*)d_in[3];
  const float* w_hh  = (const float*)d_in[4];
  const float* b_ih  = (const float*)d_in[5];
  const float* b_hh  = (const float*)d_in[6];
  const float* fc_w  = (const float*)d_in[7];
  const float* fc_b  = (const float*)d_in[8];
  const int*   adj   = (const int*)d_in[9];

  float* out      = (float*)d_out;
  float* out_pred = out;
  float* out_zbar = out + (size_t)NN * FD;
  float* out_adj  = out + (size_t)NN * FD * (1 + KK);

  char* ws = (char*)d_ws;
  size_t off = 0;
  auto take = [&](size_t b) { char* p = ws + off; off += (b + 255) & ~(size_t)255; return p; };
  unsigned int*   abits = (unsigned int*)  take((size_t)KK * NN * (NN / 32) * 4);
  float*          dinv  = (float*)         take((size_t)KK * NN * 4);
  unsigned short* xbf   = (unsigned short*)take((size_t)NN * FD * 2);
  unsigned short* gnnw  = (unsigned short*)take((size_t)KK * FD * HD * 2);
  unsigned short* wihT  = (unsigned short*)take((size_t)KK * HD * 4 * HD * 2);
  unsigned short* whhT  = (unsigned short*)take((size_t)KK * HD * 4 * HD * 2);
  unsigned short* fcwT  = (unsigned short*)take((size_t)KK * HD * FD * 2);
  float*          xw    = (float*)         take((size_t)NN * HD * 4);
  unsigned short* ybf   = (unsigned short*)take((size_t)NN * HD * 2);
  float*          zb    = (float*)         take((size_t)NN * FD * 4);   // also dpre
  unsigned short* gbf   = (unsigned short*)take((size_t)NN * HD * 2);
  float*          xg    = (float*)         take((size_t)NN * 4 * HD * 4);
  float*          hg    = (float*)         take((size_t)NN * 4 * HD * 4);
  float*          cbuf  = (float*)         take((size_t)NN * HD * 4);
  unsigned short* hbf   = (unsigned short*)take((size_t)NN * HD * 2);
  float*          dsum  = (float*)         take((size_t)NN * FD * 4);

  prep_adj<<<NN, 128, 0, stream>>>(adj, out_adj, abits, dinv);
  cvt<<<(NN * FD) / 256, 256, 0, stream>>>(x, xbf, NN * FD);
  cvt<<<(KK * FD * HD) / 256, 256, 0, stream>>>(gnn_w, gnnw, KK * FD * HD);
  cvtT<<<dim3((4 * HD * HD) / 256, 1, KK), 256, 0, stream>>>(w_ih, wihT, 4 * HD, HD);
  cvtT<<<dim3((4 * HD * HD) / 256, 1, KK), 256, 0, stream>>>(w_hh, whhT, 4 * HD, HD);
  cvtT<<<dim3((FD * HD) / 256, 1, KK), 256, 0, stream>>>(fc_w, fcwT, FD, HD);

  for (int k = 0; k < KK; ++k) {
    gemm_bf16<<<dim3(NN / BM, HD / BN), 256, 0, stream>>>(
        xbf, gnnw + (size_t)k * FD * HD, xw, NN, HD, FD);
    e1_scale<<<(NN * HD) / 256, 256, 0, stream>>>(xw, dinv + k * NN, ybf);
    gemm_adj<<<dim3(NN / BM, HD / BN), 256, 0, stream>>>(
        abits + (size_t)k * NN * (NN / 32), ybf, zb, HD, NN);
    e2_gcn<<<(NN * HD) / 256, 256, 0, stream>>>(zb, xw, dinv + k * NN, gnn_b + k * HD, gbf);
    gemm_bf16<<<dim3(NN / BM, (4 * HD) / BN), 256, 0, stream>>>(
        gbf, wihT + (size_t)k * HD * 4 * HD, xg, NN, 4 * HD, HD);
    for (int t = 0; t < TT; ++t) {
      if (t > 0)
        gemm_bf16<<<dim3(NN / BM, (4 * HD) / BN), 256, 0, stream>>>(
            hbf, whhT + (size_t)k * HD * 4 * HD, hg, NN, 4 * HD, HD);
      lstm_pw<<<(NN * HD) / 256, 256, 0, stream>>>(
          xg, hg, b_ih + k * 4 * HD, b_hh + k * 4 * HD, cbuf, hbf, t == 0 ? 1 : 0);
    }
    gemm_bf16<<<dim3(NN / BM, FD / BN), 256, 0, stream>>>(
        hbf, fcwT + (size_t)k * HD * FD, zb, NN, FD, HD);
    e3_fc<<<(NN * FD) / 256, 256, 0, stream>>>(zb, fc_b + k * FD, x, dsum, out_zbar, k);
  }
  e4_pred<<<(NN * FD) / 256, 256, 0, stream>>>(x, dsum, out_pred);
}
